// WaveletBlock_64183991272089
// MI455X (gfx1250) — matrix operand order repaired and checked by reading, not compiled, not run on hardware
//
#include <hip/hip_runtime.h>

// ---------------------------------------------------------------------------
// MI455X (gfx1250) implementation of the WaveletBlock reference.
// - All matmuls via v_wmma_f32_16x16x32_f16 (f16 operands, f32 accumulate).
// - GEMM tiles staged with GLOBAL_LOAD_ASYNC_TO_LDS_B128 (ASYNCcnt DMA),
//   double-buffered, K templated so the main loop fully unrolls.
// - Selective scan parallelized as a 3-pass chunked diagonal-linear-recurrence
//   scan (32 chunks of 128 steps): 1024-block local pass, 32-block stitch,
//   1024-block replay pass. dt-rank-8 projection + softplus fused in-loop.
// ---------------------------------------------------------------------------

typedef __attribute__((ext_vector_type(16))) _Float16 v16h;
typedef __attribute__((ext_vector_type(8)))  _Float16 v8h;
typedef __attribute__((ext_vector_type(8)))  float    v8f;

enum { EPI_F32 = 0, EPI_F16 = 1, EPI_SPLIT_XZ = 2, EPI_NCHW = 3, EPI_DUAL = 4, EPI_SIGMOID = 5 };

#define TILE_M 128
#define TILE_N 64
#define TILE_K 32

// CDNA5 async LDS DMA (ASYNCcnt-tracked): 16B per lane, no VGPR round trip.
__device__ __forceinline__ void async_ld_b128(unsigned lds_addr, const void* gaddr) {
  asm volatile("global_load_async_to_lds_b128 %0, %1, off"
               :: "v"(lds_addr), "v"(gaddr) : "memory");
}
__device__ __forceinline__ void wait_async_0() {
  asm volatile("s_wait_asynccnt 0x0" ::: "memory");
}

// ---------------------------------------------------------------------------
// WMMA GEMM: out = A(M x K f16) @ Wt(N x K f16)^T (+bias, fused epilogues).
// 256 threads = 8 waves (4Mx2N), each wave 32x32 via 2x2 16x16x32 tiles.
// Double-buffered async-to-LDS staging; weight rows pre-padded to mult of 64.
// amode=1 flips rows within each batch of 4096 (reverse scan directions).
// ---------------------------------------------------------------------------
template <int K>
__global__ void __launch_bounds__(256) wmma_gemm_kernel(
    const _Float16* __restrict__ A,
    const _Float16* __restrict__ Wt,
    const float* __restrict__ bias,
    float* __restrict__ out0f,
    _Float16* __restrict__ out1h,
    int ldc, int ncols, int epi, int amode)
{
  constexpr int NK = K / TILE_K;
  constexpr int LDA = TILE_K + 8;
  __shared__ _Float16 As[2][TILE_M][LDA];
  __shared__ _Float16 Bs[2][TILE_N][LDA];

  const int tid      = threadIdx.x;
  const int lane     = tid & 31;
  const int laneHalf = lane >> 4;
  const int l16      = lane & 15;
  const int wave     = tid >> 5;
  const int waveM    = (wave >> 1) * 32;
  const int waveN    = (wave & 1) * 32;
  const int mBase    = blockIdx.x * TILE_M;
  const int nBase    = blockIdx.y * TILE_N;

  const unsigned asAddr = (unsigned)(size_t)&As[0][0][0];
  const unsigned bsAddr = (unsigned)(size_t)&Bs[0][0][0];

  auto issue = [&](int k0, int buf) {
    unsigned aB = asAddr + (unsigned)buf * (TILE_M * LDA * 2);
    unsigned bB = bsAddr + (unsigned)buf * (TILE_N * LDA * 2);
#pragma unroll
    for (int c = 0; c < 2; ++c) {       // A tile: 128x32 = 512 16B chunks
      int idx = tid + c * 256;
      int row = idx >> 2;
      int kc  = idx & 3;
      int gm  = mBase + row;
      int bb  = gm >> 12;               // L = 4096 rows per batch
      int ll  = gm & 4095;
      if (amode) ll = 4095 - ll;
      async_ld_b128(aB + (unsigned)(row * LDA + kc * 8) * 2,
                    A + ((size_t)(bb << 12) + (size_t)ll) * K + k0 + kc * 8);
    }
    {                                   // B tile: 64x32 = 256 16B chunks
      int row = tid >> 2;
      int kc  = tid & 3;
      async_ld_b128(bB + (unsigned)(row * LDA + kc * 8) * 2,
                    Wt + (size_t)(nBase + row) * K + k0 + kc * 8);
    }
  };

  v8f acc[2][2] = {};
  issue(0, 0);

#pragma unroll
  for (int i = 0; i < NK; ++i) {
    wait_async_0();          // this wave's portion of tile i is in LDS
    __syncthreads();         // everyone's portion is in LDS; buf i&1 readable,
                             // buf (i+1)&1 no longer being read by anyone
    if (i + 1 < NK) issue((i + 1) * TILE_K, (i + 1) & 1);
    const int buf = i & 1;

    // Fragment loads per CDNA5 ISA 7.12.2 layouts.
    v16h af[2], bf[2];
#pragma unroll
    for (int t = 0; t < 2; ++t) {
      const _Float16* ap = &As[buf][waveM + 16 * t + l16][0];
      v8h a0 = *reinterpret_cast<const v8h*>(ap + 8 * laneHalf);
      v8h a1 = *reinterpret_cast<const v8h*>(ap + 16 + 8 * laneHalf);
      af[t] = __builtin_shufflevector(a0, a1, 0,1,2,3,4,5,6,7,8,9,10,11,12,13,14,15);
      const _Float16* bp = &Bs[buf][waveN + 16 * t + l16][0];
      v8h b0 = *reinterpret_cast<const v8h*>(bp + 8 * laneHalf);
      v8h b1 = *reinterpret_cast<const v8h*>(bp + 16 + 8 * laneHalf);
      bf[t] = __builtin_shufflevector(b0, b1, 0,1,2,3,4,5,6,7,8,9,10,11,12,13,14,15);
    }
#pragma unroll
    for (int im = 0; im < 2; ++im)
#pragma unroll
      for (int in_ = 0; in_ < 2; ++in_)
        acc[im][in_] = __builtin_amdgcn_wmma_f32_16x16x32_f16(
            false, af[im], false, bf[in_], (short)0, acc[im][in_], false, false);
  }

  // Epilogue: C layout — VGPR i: lanes 0-15 row i, lanes 16-31 row i+8; col = lane&15.
#pragma unroll
  for (int im = 0; im < 2; ++im) {
#pragma unroll
    for (int in_ = 0; in_ < 2; ++in_) {
#pragma unroll
      for (int i = 0; i < 8; ++i) {
        int m = mBase + waveM + 16 * im + i + 8 * laneHalf;
        int n = nBase + waveN + 16 * in_ + l16;
        if (n >= ncols) continue;
        float v  = acc[im][in_][i];
        float bv = bias ? bias[n] : 0.f;
        int bb = m >> 12;
        int ll = m & 4095;
        switch (epi) {
          case EPI_F32:
            out0f[(size_t)m * ldc + n] = v + bv;
            break;
          case EPI_F16:
            out1h[(size_t)m * ldc + n] = (_Float16)(v + bv);
            break;
          case EPI_SPLIT_XZ:  // cols<256 -> xx raw (NCHW f32), cols>=256 -> z (f16)
            if (n < 256) out0f[((size_t)bb * 256 + n) * 4096 + ll] = v;
            else         out1h[(size_t)m * 256 + (n - 256)] = (_Float16)v;
            break;
          case EPI_NCHW:
            out0f[((size_t)bb * ncols + n) * 4096 + ll] = v + bv;
            break;
          case EPI_DUAL:
            out0f[(size_t)m * ldc + n] = v;
            out1h[(size_t)m * ldc + n] = (_Float16)v;
            break;
          case EPI_SIGMOID:
            out0f[(size_t)m * ldc + n] = 1.f / (1.f + __expf(-(v + bv)));
            break;
        }
      }
    }
  }
}

// ---------------------------------------------------------------------------
// Haar wavelet split: x(8,64,128,128) -> xll16 (b,h,w,64) + xhf16 (b,h,w,192)
// ---------------------------------------------------------------------------
__global__ void __launch_bounds__(256) wavelet_kernel(
    const float* __restrict__ x, _Float16* __restrict__ xll, _Float16* __restrict__ xhf)
{
  int t = blockIdx.x * 256 + threadIdx.x;   // ((b*64+h)*64+w)*64+c
  int c = t & 63;
  int w = (t >> 6) & 63;
  int h = (t >> 12) & 63;
  int b = t >> 18;
  const float* xb = x + ((size_t)(b * 64 + c)) * 16384;
  float x1 = xb[(2 * h) * 128 + 2 * w] * 0.5f;
  float x3 = xb[(2 * h) * 128 + 2 * w + 1] * 0.5f;
  float x2 = xb[(2 * h + 1) * 128 + 2 * w] * 0.5f;
  float x4 = xb[(2 * h + 1) * 128 + 2 * w + 1] * 0.5f;
  xll[t] = (_Float16)(x1 + x2 + x3 + x4);
  size_t hb = ((size_t)((b * 64 + h) * 64 + w)) * 192;
  xhf[hb + c]       = (_Float16)(-x1 - x2 + x3 + x4);
  xhf[hb + 64 + c]  = (_Float16)(-x1 + x2 - x3 + x4);
  xhf[hb + 128 + c] = (_Float16)(x1 - x2 - x3 + x4);
}

// ---------------------------------------------------------------------------
// Depthwise 3x3 + bias + SiLU on xx (NCHW f32, 256ch), writing both scan
// layouts: xsT[0][b][h*64+w][d] and xsT[1][b][w*64+h][d] (f16).
// ---------------------------------------------------------------------------
__global__ void __launch_bounds__(256) dwconv_low_kernel(
    const float* __restrict__ xin, const float* __restrict__ cw,
    const float* __restrict__ cb, _Float16* __restrict__ xsT)
{
  int t = blockIdx.x * 256 + threadIdx.x;   // (b*4096+l)*256+d
  int d = t & 255;
  int l = (t >> 8) & 4095;
  int b = t >> 20;
  int h = l >> 6, w = l & 63;
  const float* xc = xin + ((size_t)(b * 256 + d)) * 4096;
  const float* wv = cw + d * 9;
  float acc = cb[d];
#pragma unroll
  for (int dh = -1; dh <= 1; ++dh) {
    int hh = h + dh;
    if (hh < 0 || hh > 63) continue;
#pragma unroll
    for (int dw_ = -1; dw_ <= 1; ++dw_) {
      int ww = w + dw_;
      if (ww < 0 || ww > 63) continue;
      acc += wv[(dh + 1) * 3 + (dw_ + 1)] * xc[hh * 64 + ww];
    }
  }
  float s = acc / (1.f + __expf(-acc));
  xsT[((size_t)b * 4096 + l) * 256 + d] = (_Float16)s;
  xsT[((size_t)(8 + b) * 4096 + (w * 64 + h)) * 256 + d] = (_Float16)s;
}

// High path depthwise 3x3 + bias + SiLU: fhf0 (NCHW f32,128ch) -> (b,hw,128) f16
__global__ void __launch_bounds__(256) dwconv_high_kernel(
    const float* __restrict__ xin, const float* __restrict__ cw,
    const float* __restrict__ cb, _Float16* __restrict__ outh)
{
  int t = blockIdx.x * 256 + threadIdx.x;   // (b*4096+l)*128+c
  int c = t & 127;
  int l = (t >> 7) & 4095;
  int b = t >> 19;
  int h = l >> 6, w = l & 63;
  const float* xc = xin + ((size_t)(b * 128 + c)) * 4096;
  const float* wv = cw + c * 9;
  float acc = cb[c];
#pragma unroll
  for (int dh = -1; dh <= 1; ++dh) {
    int hh = h + dh;
    if (hh < 0 || hh > 63) continue;
#pragma unroll
    for (int dw_ = -1; dw_ <= 1; ++dw_) {
      int ww = w + dw_;
      if (ww < 0 || ww > 63) continue;
      acc += wv[(dh + 1) * 3 + (dw_ + 1)] * xc[hh * 64 + ww];
    }
  }
  outh[t] = (_Float16)(acc / (1.f + __expf(-acc)));
}

__global__ void zero_f32_kernel(float* __restrict__ p, int n)
{
  int t = blockIdx.x * 256 + threadIdx.x;
  if (t < n) p[t] = 0.f;
}

// ---------------------------------------------------------------------------
// Chunked selective scan. h is a diagonal linear recurrence, so per chunk
// h_out = P*h_in + q with P = prod(dA). 32 chunks x 128 steps.
//   pass1 (1024 blocks): per-chunk P, q (h starting from 0)
//   pass2 (  32 blocks): sequential stitch -> h_in per chunk
//   pass3 (1024 blocks): replay chunk from h_in, emit y (atomic merge of the
//                        4 scan directions with flip/transpose remaps)
// ---------------------------------------------------------------------------
#define NCH 32
#define CLEN 128

__global__ void __launch_bounds__(256) scan_part1_kernel(
    const _Float16* __restrict__ xsT, const float* __restrict__ xdbl,
    const float* __restrict__ A_logs, const float* __restrict__ dtw,
    const float* __restrict__ dtb,
    float* __restrict__ gP, float* __restrict__ gQ)
{
  int bk = blockIdx.x >> 5;     // (b*4+k)
  int ch = blockIdx.x & 31;
  int b = bk >> 2, k = bk & 3;
  int d = threadIdx.x;
  int kd = k * 256 + d;

  float An[16], w8[8];
#pragma unroll
  for (int n = 0; n < 16; ++n) An[n] = -__expf(A_logs[kd * 16 + n]);
#pragma unroll
  for (int r = 0; r < 8; ++r) w8[r] = dtw[kd * 8 + r];
  const float bdt = dtb[kd];
  float h[16], P[16];
#pragma unroll
  for (int n = 0; n < 16; ++n) { h[n] = 0.f; P[n] = 1.f; }

  const _Float16* uB = xsT + (((size_t)(k & 1) * 8 + b) * 4096) * 256 + d;
  const float* xd = xdbl + ((size_t)(k * 8 + b)) * 4096 * 40;
  const bool fl = (k >= 2);
  __shared__ float sm[40];

  int l0 = ch * CLEN;
  for (int t = 0; t < CLEN; ++t) {
    int l = l0 + t;
    __syncthreads();
    if (threadIdx.x < 40) sm[threadIdx.x] = xd[(size_t)l * 40 + threadIdx.x];
    __syncthreads();
    int sl = fl ? (4095 - l) : l;
    float u = (float)uB[(size_t)sl * 256];
    float dt = bdt;
#pragma unroll
    for (int r = 0; r < 8; ++r) dt += sm[r] * w8[r];
    dt = (dt > 20.f) ? dt : __logf(1.f + __expf(dt));
#pragma unroll
    for (int n = 0; n < 16; ++n) {
      float dA = __expf(dt * An[n]);
      h[n] = dA * h[n] + (dt * sm[8 + n]) * u;
      P[n] *= dA;
    }
  }
  size_t o = (((size_t)bk * NCH + ch) * 256 + d) * 16;
#pragma unroll
  for (int n = 0; n < 16; ++n) { gP[o + n] = P[n]; gQ[o + n] = h[n]; }
}

__global__ void __launch_bounds__(256) scan_part2_kernel(
    const float* __restrict__ gP, const float* __restrict__ gQ,
    float* __restrict__ gH)
{
  int bk = blockIdx.x;
  int d = threadIdx.x;
  float h[16];
#pragma unroll
  for (int n = 0; n < 16; ++n) h[n] = 0.f;
  for (int ch = 0; ch < NCH; ++ch) {
    size_t o = (((size_t)bk * NCH + ch) * 256 + d) * 16;
#pragma unroll
    for (int n = 0; n < 16; ++n) gH[o + n] = h[n];
#pragma unroll
    for (int n = 0; n < 16; ++n) h[n] = gP[o + n] * h[n] + gQ[o + n];
  }
}

__global__ void __launch_bounds__(256) scan_part3_kernel(
    const _Float16* __restrict__ xsT, const float* __restrict__ xdbl,
    const float* __restrict__ A_logs, const float* __restrict__ dtw,
    const float* __restrict__ dtb, const float* __restrict__ Dsp,
    const float* __restrict__ gH, float* __restrict__ ysum)
{
  int bk = blockIdx.x >> 5;
  int ch = blockIdx.x & 31;
  int b = bk >> 2, k = bk & 3;
  int d = threadIdx.x;
  int kd = k * 256 + d;

  float An[16], w8[8];
#pragma unroll
  for (int n = 0; n < 16; ++n) An[n] = -__expf(A_logs[kd * 16 + n]);
#pragma unroll
  for (int r = 0; r < 8; ++r) w8[r] = dtw[kd * 8 + r];
  const float bdt = dtb[kd];
  const float dsv = Dsp[kd];

  float h[16];
  size_t o = (((size_t)bk * NCH + ch) * 256 + d) * 16;
#pragma unroll
  for (int n = 0; n < 16; ++n) h[n] = gH[o + n];

  const _Float16* uB = xsT + (((size_t)(k & 1) * 8 + b) * 4096) * 256 + d;
  const float* xd = xdbl + ((size_t)(k * 8 + b)) * 4096 * 40;
  float* ys = ysum + (size_t)b * 4096 * 256 + d;
  const bool fl = (k >= 2);
  __shared__ float sm[40];

  int l0 = ch * CLEN;
  for (int t = 0; t < CLEN; ++t) {
    int l = l0 + t;
    __syncthreads();
    if (threadIdx.x < 40) sm[threadIdx.x] = xd[(size_t)l * 40 + threadIdx.x];
    __syncthreads();
    int sl = fl ? (4095 - l) : l;
    float u = (float)uB[(size_t)sl * 256];
    float dt = bdt;
#pragma unroll
    for (int r = 0; r < 8; ++r) dt += sm[r] * w8[r];
    dt = (dt > 20.f) ? dt : __logf(1.f + __expf(dt));
    float y = 0.f;
#pragma unroll
    for (int n = 0; n < 16; ++n) {
      float dA = __expf(dt * An[n]);
      h[n] = dA * h[n] + (dt * sm[8 + n]) * u;
      y += h[n] * sm[24 + n];
    }
    y += dsv * u;
    int lp;
    if (k == 0)      lp = l;
    else if (k == 1) lp = ((l & 63) << 6) | (l >> 6);
    else if (k == 2) lp = 4095 - l;
    else { int l2 = 4095 - l; lp = ((l2 & 63) << 6) | (l2 >> 6); }
    atomicAdd(&ys[(size_t)lp * 256], y);
  }
}

// ---------------------------------------------------------------------------
// LayerNorm(d=256) * silu(z) -> f16 for the out-proj GEMM.
// ---------------------------------------------------------------------------
__global__ void __launch_bounds__(256) ln_gate_kernel(
    const float* __restrict__ ysum, const _Float16* __restrict__ z16,
    const float* __restrict__ nw, const float* __restrict__ nb,
    _Float16* __restrict__ outh)
{
  int row = blockIdx.x;
  int d = threadIdx.x;
  size_t base = (size_t)row * 256;
  float v = ysum[base + d];
  __shared__ float red[256];
  red[d] = v;
  __syncthreads();
  for (int s = 128; s > 0; s >>= 1) { if (d < s) red[d] += red[d + s]; __syncthreads(); }
  float mu = red[0] * (1.f / 256.f);
  __syncthreads();
  float df = v - mu;
  red[d] = df * df;
  __syncthreads();
  for (int s = 128; s > 0; s >>= 1) { if (d < s) red[d] += red[d + s]; __syncthreads(); }
  float var = red[0] * (1.f / 256.f);
  float zv = (float)z16[base + d];
  float sz = zv / (1.f + __expf(-zv));
  float r = (df * rsqrtf(var + 1e-5f)) * nw[d] + nb[d];
  outh[base + d] = (_Float16)(r * sz);
}

// out[b,o,h,w] = feat_ll + feat_hf * gate  (gate already sigmoid'd in epilogue)
__global__ void __launch_bounds__(256) final_kernel(
    const float* __restrict__ fll, const float* __restrict__ fhf,
    const float* __restrict__ g, float* __restrict__ out)
{
  int t = blockIdx.x * 256 + threadIdx.x;  // ((b*128+o)*4096)+l  (NCHW flat)
  int l = t & 4095;
  int o = (t >> 12) & 127;
  int b = t >> 19;
  size_t r = ((size_t)b * 4096 + l) * 128 + o;
  out[t] = fll[r] + fhf[r] * g[r];
}

// f32 -> f16 weight conversion
__global__ void cvt_f16_kernel(const float* __restrict__ s, _Float16* __restrict__ d, int n)
{
  int t = blockIdx.x * 256 + threadIdx.x;
  if (t < n) d[t] = (_Float16)s[t];
}

// x_proj_w (4,40,256) -> f16 padded to (4,64,256), rows 40..63 zero
__global__ void cvt_xproj_kernel(const float* __restrict__ s, _Float16* __restrict__ d, int n)
{
  int t = blockIdx.x * 256 + threadIdx.x;
  if (t >= n) return;
  int dd = t & 255;
  int c = (t >> 8) & 63;
  int k = t >> 14;
  d[t] = (c < 40) ? (_Float16)s[((k * 40 + c) << 8) + dd] : (_Float16)0.f;
}

// ---------------------------------------------------------------------------
extern "C" void kernel_launch(void* const* d_in, const int* in_sizes, int n_in,
                              void* d_out, int out_size, void* d_ws, size_t ws_size,
                              hipStream_t stream)
{
  (void)in_sizes; (void)n_in; (void)out_size; (void)ws_size;
  const float* x           = (const float*)d_in[0];
  const float* proj_low_w  = (const float*)d_in[1];
  const float* proj_low_b  = (const float*)d_in[2];
  const float* proj_high_w = (const float*)d_in[3];
  const float* proj_high_b = (const float*)d_in[4];
  const float* in_proj_w   = (const float*)d_in[5];
  const float* conv2d_w    = (const float*)d_in[6];
  const float* conv2d_b    = (const float*)d_in[7];
  const float* x_proj_w    = (const float*)d_in[8];
  const float* dt_projs_w  = (const float*)d_in[9];
  const float* dt_projs_b  = (const float*)d_in[10];
  const float* A_logs      = (const float*)d_in[11];
  const float* Ds          = (const float*)d_in[12];
  const float* out_norm_w  = (const float*)d_in[13];
  const float* out_norm_b  = (const float*)d_in[14];
  const float* ss_out_w    = (const float*)d_in[15];
  const float* dw_w        = (const float*)d_in[16];
  const float* dw_b        = (const float*)d_in[17];
  const float* pw_w        = (const float*)d_in[18];
  const float* pw_b        = (const float*)d_in[19];
  const float* gate_w      = (const float*)d_in[20];
  const float* gate_b      = (const float*)d_in[21];
  float* out = (float*)d_out;

  const int BHW = 32768;  // 8 * 64 * 64

  char* wsb = (char*)d_ws;
  size_t off = 0;
  auto alloc = [&](size_t bytes) -> void* {
    void* p = wsb + off;
    off = (off + bytes + 255) & ~(size_t)255;
    return p;
  };

  _Float16* xll16   = (_Float16*)alloc((size_t)BHW * 64 * 2);
  _Float16* xhf16   = (_Float16*)alloc((size_t)BHW * 192 * 2);
  _Float16* fll0_16 = (_Float16*)alloc((size_t)BHW * 128 * 2);
  _Float16* z16     = (_Float16*)alloc((size_t)BHW * 256 * 2);
  float*    xxc     = (float*)   alloc((size_t)8 * 256 * 4096 * 4);
  _Float16* xsT16   = (_Float16*)alloc((size_t)2 * 8 * 4096 * 256 * 2);
  float*    xdbl    = (float*)   alloc((size_t)4 * 8 * 4096 * 40 * 4);
  float*    ysum    = (float*)   alloc((size_t)BHW * 256 * 4);
  _Float16* ynorm16 = (_Float16*)alloc((size_t)BHW * 256 * 2);
  float*    fll_ss  = (float*)   alloc((size_t)BHW * 128 * 4);
  _Float16* fllss16 = (_Float16*)alloc((size_t)BHW * 128 * 2);
  float*    fhf0    = (float*)   alloc((size_t)8 * 128 * 4096 * 4);
  _Float16* fhf1_16 = (_Float16*)alloc((size_t)BHW * 128 * 2);
  float*    fhf2    = (float*)   alloc((size_t)BHW * 128 * 4);
  float*    gsig    = (float*)   alloc((size_t)BHW * 128 * 4);
  float*    gP      = (float*)   alloc((size_t)32 * NCH * 256 * 16 * 4);
  float*    gQ      = (float*)   alloc((size_t)32 * NCH * 256 * 16 * 4);
  float*    gH      = (float*)   alloc((size_t)32 * NCH * 256 * 16 * 4);
  _Float16* plw16   = (_Float16*)alloc(128 * 64 * 2);
  _Float16* phw16   = (_Float16*)alloc(128 * 192 * 2);
  _Float16* ipw16   = (_Float16*)alloc(512 * 128 * 2);
  _Float16* xpj16   = (_Float16*)alloc(4 * 64 * 256 * 2);
  _Float16* sow16   = (_Float16*)alloc(128 * 256 * 2);
  _Float16* pww16   = (_Float16*)alloc(128 * 128 * 2);
  _Float16* gtw16   = (_Float16*)alloc(128 * 128 * 2);

  auto cvt = [&](const float* s, _Float16* d, int n) {
    cvt_f16_kernel<<<(n + 255) / 256, 256, 0, stream>>>(s, d, n);
  };
  auto gemm = [&](const _Float16* A, const _Float16* Wt, const float* bias,
                  float* o0, _Float16* o1, int M, int N, int K,
                  int ldc, int epi, int amode) {
    dim3 g(M / TILE_M, (N + TILE_N - 1) / TILE_N);
    switch (K) {
      case 64:  wmma_gemm_kernel<64> <<<g, 256, 0, stream>>>(A, Wt, bias, o0, o1, ldc, N, epi, amode); break;
      case 128: wmma_gemm_kernel<128><<<g, 256, 0, stream>>>(A, Wt, bias, o0, o1, ldc, N, epi, amode); break;
      case 192: wmma_gemm_kernel<192><<<g, 256, 0, stream>>>(A, Wt, bias, o0, o1, ldc, N, epi, amode); break;
      default:  wmma_gemm_kernel<256><<<g, 256, 0, stream>>>(A, Wt, bias, o0, o1, ldc, N, epi, amode); break;
    }
  };

  // 0) weights -> f16 (x_proj padded to 64 rows so B tiles never need guards)
  cvt(proj_low_w,  plw16, 128 * 64);
  cvt(proj_high_w, phw16, 128 * 192);
  cvt(in_proj_w,   ipw16, 512 * 128);
  cvt(ss_out_w,    sow16, 128 * 256);
  cvt(pw_w,        pww16, 128 * 128);
  cvt(gate_w,      gtw16, 128 * 128);
  cvt_xproj_kernel<<<(4 * 64 * 256) / 256, 256, 0, stream>>>(x_proj_w, xpj16, 4 * 64 * 256);

  // 1) Haar wavelet split
  wavelet_kernel<<<(BHW * 64) / 256, 256, 0, stream>>>(x, xll16, xhf16);

  // 2) low projection (bias, f16 out)
  gemm(xll16, plw16, proj_low_b, nullptr, fll0_16, BHW, 128, 64, 128, EPI_F16, 0);

  // 3) in_proj: split into xx (NCHW f32 for conv) and z (f16)
  gemm(fll0_16, ipw16, nullptr, xxc, z16, BHW, 512, 128, 0, EPI_SPLIT_XZ, 0);

  // 4) depthwise conv + SiLU -> both scan layouts (f16)
  dwconv_low_kernel<<<(8 * 4096 * 256) / 256, 256, 0, stream>>>(xxc, conv2d_w, conv2d_b, xsT16);

  // 5) x-proj GEMMs per direction (flips realized in the A loader)
  for (int k = 0; k < 4; ++k) {
    const _Float16* Ak = xsT16 + (size_t)(k & 1) * 8 * 4096 * 256;
    gemm(Ak, xpj16 + (size_t)k * 64 * 256, nullptr,
         xdbl + (size_t)k * 8 * 4096 * 40, nullptr,
         BHW, 40, 256, 40, EPI_F32, (k >= 2) ? 1 : 0);
  }

  // 6) chunked selective scan (3 passes), directions merged via f32 atomics
  zero_f32_kernel<<<(BHW * 256) / 256, 256, 0, stream>>>(ysum, BHW * 256);
  scan_part1_kernel<<<32 * NCH, 256, 0, stream>>>(xsT16, xdbl, A_logs, dt_projs_w,
                                                  dt_projs_b, gP, gQ);
  scan_part2_kernel<<<32, 256, 0, stream>>>(gP, gQ, gH);
  scan_part3_kernel<<<32 * NCH, 256, 0, stream>>>(xsT16, xdbl, A_logs, dt_projs_w,
                                                  dt_projs_b, Ds, gH, ysum);

  // 7) layernorm * silu(z) -> f16
  ln_gate_kernel<<<BHW, 256, 0, stream>>>(ysum, z16, out_norm_w, out_norm_b, ynorm16);

  // 8) ss out projection (dual f32 + f16 output)
  gemm(ynorm16, sow16, nullptr, fll_ss, fllss16, BHW, 128, 256, 128, EPI_DUAL, 0);

  // 9) high path: proj(+bias, NCHW) -> dw conv+SiLU -> pointwise GEMM(+bias)
  gemm(xhf16, phw16, proj_high_b, fhf0, nullptr, BHW, 128, 192, 128, EPI_NCHW, 0);
  dwconv_high_kernel<<<(8 * 4096 * 128) / 256, 256, 0, stream>>>(fhf0, dw_w, dw_b, fhf1_16);
  gemm(fhf1_16, pww16, pw_b, fhf2, nullptr, BHW, 128, 128, 128, EPI_F32, 0);

  // 10) gate GEMM with fused sigmoid
  gemm(fllss16, gtw16, gate_b, gsig, nullptr, BHW, 128, 128, 128, EPI_SIGMOID, 0);

  // 11) combine + NCHW transpose
  final_kernel<<<(8 * 128 * 4096) / 256, 256, 0, stream>>>(fll_ss, fhf2, gsig, out);
}
